// _ResidualMessageLayer_34849364640430
// MI455X (gfx1250) — compile-verified, matrix operand order checked by reading, not run on hardware
//
#include <hip/hip_runtime.h>

// ---------------------------------------------------------------------------
// Fused residual message-passing layer for gfx1250 (MI455X).
//   messages = silu([nf[src], nf[dst], ef, c[src], c[dst]] @ W1 + b1) @ W2 + b2
//   agg      = segment_sum(messages, dst)
//   upd      = silu([nf, agg, c] @ U1 + ub1) @ U2 + ub2
//   out      = layernorm(nf + upd)
// All GEMMs run on v_wmma_f32_16x16x32_f16 (wave32, 16x16 tiles, K=32 steps).
// M=32-blocked edge kernel; ALL packed weights are staged in the 320KB WGP
// LDS once per block, so the WMMA loop feeds from ds_load_b128 (short DScnt
// latency) instead of streaming 112KB/wave from L2.
// ---------------------------------------------------------------------------

typedef _Float16 v16h __attribute__((ext_vector_type(16)));
typedef _Float16 v8h  __attribute__((ext_vector_type(8)));
typedef float    v8f  __attribute__((ext_vector_type(8)));

union AFrag { v16h v; v8h q[2]; _Float16 h[16]; };

#define HID 128

// sizes in halves of the packed weight buffers
#define W1_H 40960   // 10 ktiles * 8 ntiles * 512
#define W2_H 16384   //  4 ktiles * 8 ntiles * 512
#define U1_H 36864   //  9 ktiles * 8 ntiles * 512
#define U2_H 16384

// --- weight packing: fp32 [K,128] -> f16 WMMA B-fragments, K zero-padded ----
// Fragment element order: ((kt*8+nt)*32 + lane)*16 + h   (one v16h per lane).
// lane<16 : n=nt*16+lane,    h0..7 -> k=kt*32+0..7,  h8..15 -> k=kt*32+16..23
// lane>=16: n=nt*16+lane-16, h0..7 -> k=kt*32+8..15, h8..15 -> k=kt*32+24..31
__global__ __launch_bounds__(256) void pack_w(const float* __restrict__ W,
                                              _Float16* __restrict__ out,
                                              int k_orig, int total) {
    int tid = blockIdx.x * 256 + threadIdx.x;
    if (tid >= total) return;
    int h     = tid & 15;
    int lane  = (tid >> 4) & 31;
    int tileI = tid >> 9;
    int kt = tileI >> 3, nt = tileI & 7;
    int n  = nt * 16 + (lane & 15);
    int kb = kt * 32 + ((lane >> 4) ? 8 : 0);
    int k  = kb + (h < 8 ? h : h + 8);
    float val = (k < k_orig) ? W[(long long)k * HID + n] : 0.0f;
    out[tid] = (_Float16)val;
}

__global__ __launch_bounds__(256) void zero_f32(float* __restrict__ p, long long n) {
    long long i = (long long)blockIdx.x * 256 + threadIdx.x;
    if (i < n) p[i] = 0.0f;
}

// --- 8 contiguous f32 -> 8 f16 ---------------------------------------------
__device__ __forceinline__ void cvt8(_Float16* o, const float* __restrict__ p) {
    const float4* q = (const float4*)p;
    float4 a = q[0], b = q[1];
    o[0] = (_Float16)a.x; o[1] = (_Float16)a.y; o[2] = (_Float16)a.z; o[3] = (_Float16)a.w;
    o[4] = (_Float16)b.x; o[5] = (_Float16)b.y; o[6] = (_Float16)b.z; o[7] = (_Float16)b.w;
}

// edge_input[k]: k<128 nf[src], k<256 nf[dst], k<288 ef[e], 288 c[src], 289 c[dst]
__device__ __forceinline__ void edge_run8(_Float16* o, int kb,
                                          const float* __restrict__ nf,
                                          const float* __restrict__ ef,
                                          const float* __restrict__ coord,
                                          long long e, long long s, long long d) {
    if (kb < 128)       cvt8(o, nf + s * HID + kb);
    else if (kb < 256)  cvt8(o, nf + d * HID + (kb - 128));
    else if (kb < 288)  cvt8(o, ef + e * 32 + (kb - 256));
    else if (kb == 288) {
        o[0] = (_Float16)coord[s]; o[1] = (_Float16)coord[d];
        #pragma unroll
        for (int i = 2; i < 8; ++i) o[i] = (_Float16)0.0f;
    } else {
        #pragma unroll
        for (int i = 0; i < 8; ++i) o[i] = (_Float16)0.0f;
    }
}

// update_input[k]: k<128 nf[n], k<256 agg[n], 256 c[n]
__device__ __forceinline__ void node_run8(_Float16* o, int kb,
                                          const float* __restrict__ nf,
                                          const float* __restrict__ agg,
                                          const float* __restrict__ coord,
                                          long long node) {
    if (kb < 128)       cvt8(o, nf + node * HID + kb);
    else if (kb < 256)  cvt8(o, agg + node * HID + (kb - 128));
    else if (kb == 256) {
        o[0] = (_Float16)coord[node];
        #pragma unroll
        for (int i = 1; i < 8; ++i) o[i] = (_Float16)0.0f;
    } else {
        #pragma unroll
        for (int i = 0; i < 8; ++i) o[i] = (_Float16)0.0f;
    }
}

// --- edge kernel: message MLP + atomic scatter, 32 edges per wave -----------
// LDS: [0 .. 57343] halves = W1|W2 fragments, then 4 * 32*128 halves h-staging
__global__ __launch_bounds__(128) void edge_msg(
    const float* __restrict__ nf, const long long* __restrict__ ei,
    const float* __restrict__ ef, const float* __restrict__ coord,
    const _Float16* __restrict__ W1p, const float* __restrict__ b1,
    const _Float16* __restrict__ W2p, const float* __restrict__ b2,
    float* __restrict__ agg, int n_edges) {
    __shared__ __align__(16) _Float16 smem[W1_H + W2_H + 4 * 32 * HID]; // 144KB

    // ---- cooperative stage of packed weights into LDS (before any exit) ----
    {
        uint4*       lw = (uint4*)smem;
        const uint4* g1 = (const uint4*)W1p;   // 5120 uint4
        const uint4* g2 = (const uint4*)W2p;   // 2048 uint4
        for (int i = threadIdx.x; i < W1_H / 8; i += 128) lw[i] = g1[i];
        for (int i = threadIdx.x; i < W2_H / 8; i += 128)
            lw[W1_H / 8 + i] = g2[i];
    }
    __syncthreads();

    const int lane = threadIdx.x & 31;
    const int wave = threadIdx.x >> 5;
    const int r    = lane & 15;
    const int hi   = lane >> 4;
    const long long base = (long long)(blockIdx.x * 4 + wave) * 32;
    if (base >= n_edges) return;  // wave-uniform exit

    long long e0 = base + r;       if (e0 >= n_edges) e0 = n_edges - 1;
    long long e1 = base + 16 + r;  if (e1 >= n_edges) e1 = n_edges - 1;
    const long long s0 = ei[e0], d0 = ei[e0 + n_edges];
    const long long s1 = ei[e1], d1 = ei[e1 + n_edges];

    const v16h* W1t = (const v16h*)smem;            // 2560 fragments
    const v16h* W2t = (const v16h*)(smem + W1_H);   // 1024 fragments
    _Float16*   hbuf = smem + W1_H + W2_H + wave * 32 * HID;

    // ---- GEMM1: [32,320] x [320,128], accum pre-loaded with bias ----
    v8f acc[2][8];
    #pragma unroll
    for (int nt = 0; nt < 8; ++nt) {
        float bv = b1[nt * 16 + r];
        #pragma unroll
        for (int v = 0; v < 8; ++v) { acc[0][nt][v] = bv; acc[1][nt][v] = bv; }
    }
    #pragma unroll
    for (int kt = 0; kt < 10; ++kt) {
        AFrag a0, a1;
        int kb0 = kt * 32 + hi * 8;
        edge_run8(a0.h,     kb0,      nf, ef, coord, e0, s0, d0);
        edge_run8(a0.h + 8, kb0 + 16, nf, ef, coord, e0, s0, d0);
        edge_run8(a1.h,     kb0,      nf, ef, coord, e1, s1, d1);
        edge_run8(a1.h + 8, kb0 + 16, nf, ef, coord, e1, s1, d1);
        #pragma unroll
        for (int nt = 0; nt < 8; ++nt) {
            v16h b = W1t[(kt * 8 + nt) * 32 + lane];
            acc[0][nt] = __builtin_amdgcn_wmma_f32_16x16x32_f16(
                false, a0.v, false, b, (short)0, acc[0][nt], false, false);
            acc[1][nt] = __builtin_amdgcn_wmma_f32_16x16x32_f16(
                false, a1.v, false, b, (short)0, acc[1][nt], false, false);
        }
    }

    // ---- SiLU, stage h (f16) in wave-private LDS (row-major 32x128) ----
    #pragma unroll
    for (int rt = 0; rt < 2; ++rt)
        #pragma unroll
        for (int nt = 0; nt < 8; ++nt)
            #pragma unroll
            for (int v = 0; v < 8; ++v) {
                float x  = acc[rt][nt][v];
                float sv = x * (1.0f / (1.0f + __expf(-x)));
                hbuf[(rt * 16 + v + 8 * hi) * HID + nt * 16 + r] = (_Float16)sv;
            }
    __syncthreads();  // ensure DS stores visible before fragment reloads

    // ---- GEMM2: [32,128] x [128,128], accum pre-loaded with bias ----
    v8f acc2[2][8];
    #pragma unroll
    for (int nt = 0; nt < 8; ++nt) {
        float bv = b2[nt * 16 + r];
        #pragma unroll
        for (int v = 0; v < 8; ++v) { acc2[0][nt][v] = bv; acc2[1][nt][v] = bv; }
    }
    #pragma unroll
    for (int kt = 0; kt < 4; ++kt) {
        AFrag a0, a1;
        int kb0 = kt * 32 + hi * 8;
        a0.q[0] = *(const v8h*)&hbuf[r * HID + kb0];
        a0.q[1] = *(const v8h*)&hbuf[r * HID + kb0 + 16];
        a1.q[0] = *(const v8h*)&hbuf[(16 + r) * HID + kb0];
        a1.q[1] = *(const v8h*)&hbuf[(16 + r) * HID + kb0 + 16];
        #pragma unroll
        for (int nt = 0; nt < 8; ++nt) {
            v16h b = W2t[(kt * 8 + nt) * 32 + lane];
            acc2[0][nt] = __builtin_amdgcn_wmma_f32_16x16x32_f16(
                false, a0.v, false, b, (short)0, acc2[0][nt], false, false);
            acc2[1][nt] = __builtin_amdgcn_wmma_f32_16x16x32_f16(
                false, a1.v, false, b, (short)0, acc2[1][nt], false, false);
        }
    }

    // ---- scatter into L2-resident aggregate ----
    #pragma unroll
    for (int rt = 0; rt < 2; ++rt)
        #pragma unroll
        for (int v = 0; v < 8; ++v) {
            long long ee = base + rt * 16 + v + 8 * hi;
            if (ee < n_edges) {
                long long dn = ei[ee + n_edges];
                float* dstp = agg + dn * HID + r;
                #pragma unroll
                for (int nt = 0; nt < 8; ++nt)
                    unsafeAtomicAdd(dstp + nt * 16, acc2[rt][nt][v]);
            }
        }
}

// --- node kernel: update MLP + residual + layernorm -------------------------
// LDS: U1|U2 fragments, then 4 * 16*128 halves h, plus f32 out staging
__global__ __launch_bounds__(128) void node_upd(
    const float* __restrict__ nf, const float* __restrict__ agg,
    const float* __restrict__ coord,
    const _Float16* __restrict__ U1p, const float* __restrict__ b1,
    const _Float16* __restrict__ U2p, const float* __restrict__ b2,
    const float* __restrict__ gamma, const float* __restrict__ beta,
    float* __restrict__ out, int n_nodes) {
    __shared__ __align__(16) _Float16 smem[U1_H + U2_H + 4 * 16 * HID]; // 120KB
    __shared__ float lds_o[4][16 * HID];                                //  32KB

    // ---- cooperative stage of packed weights into LDS (before any exit) ----
    {
        uint4*       lw = (uint4*)smem;
        const uint4* g1 = (const uint4*)U1p;   // 4608 uint4
        const uint4* g2 = (const uint4*)U2p;   // 2048 uint4
        for (int i = threadIdx.x; i < U1_H / 8; i += 128) lw[i] = g1[i];
        for (int i = threadIdx.x; i < U2_H / 8; i += 128)
            lw[U1_H / 8 + i] = g2[i];
    }
    __syncthreads();

    const int lane = threadIdx.x & 31;
    const int wave = threadIdx.x >> 5;
    const int r    = lane & 15;
    const int hi   = lane >> 4;
    const long long base = (long long)(blockIdx.x * 4 + wave) * 16;
    if (base >= n_nodes) return;  // wave-uniform exit

    long long node = base + r; if (node >= n_nodes) node = n_nodes - 1;

    const v16h* U1t = (const v16h*)smem;
    const v16h* U2t = (const v16h*)(smem + U1_H);
    _Float16*   hbuf = smem + U1_H + U2_H + wave * 16 * HID;

    // ---- GEMM1: [16,288] x [288,128] ----
    v8f acc[8];
    #pragma unroll
    for (int nt = 0; nt < 8; ++nt) {
        float bv = b1[nt * 16 + r];
        #pragma unroll
        for (int v = 0; v < 8; ++v) acc[nt][v] = bv;
    }
    #pragma unroll
    for (int kt = 0; kt < 9; ++kt) {
        AFrag a;
        int kb0 = kt * 32 + hi * 8;
        node_run8(a.h,     kb0,      nf, agg, coord, node);
        node_run8(a.h + 8, kb0 + 16, nf, agg, coord, node);
        #pragma unroll
        for (int nt = 0; nt < 8; ++nt) {
            v16h b = U1t[(kt * 8 + nt) * 32 + lane];
            acc[nt] = __builtin_amdgcn_wmma_f32_16x16x32_f16(
                false, a.v, false, b, (short)0, acc[nt], false, false);
        }
    }

    #pragma unroll
    for (int nt = 0; nt < 8; ++nt)
        #pragma unroll
        for (int v = 0; v < 8; ++v) {
            float x  = acc[nt][v];
            float sv = x * (1.0f / (1.0f + __expf(-x)));
            hbuf[(v + 8 * hi) * HID + nt * 16 + r] = (_Float16)sv;
        }
    __syncthreads();

    // ---- GEMM2: [16,128] x [128,128] ----
    v8f acc2[8];
    #pragma unroll
    for (int nt = 0; nt < 8; ++nt) {
        float bv = b2[nt * 16 + r];
        #pragma unroll
        for (int v = 0; v < 8; ++v) acc2[nt][v] = bv;
    }
    #pragma unroll
    for (int kt = 0; kt < 4; ++kt) {
        AFrag a;
        int kb0 = kt * 32 + hi * 8;
        a.q[0] = *(const v8h*)&hbuf[r * HID + kb0];
        a.q[1] = *(const v8h*)&hbuf[r * HID + kb0 + 16];
        #pragma unroll
        for (int nt = 0; nt < 8; ++nt) {
            v16h b = U2t[(kt * 8 + nt) * 32 + lane];
            acc2[nt] = __builtin_amdgcn_wmma_f32_16x16x32_f16(
                false, a.v, false, b, (short)0, acc2[nt], false, false);
        }
    }

    // ---- residual into LDS (f32) ----
    float* obuf = lds_o[wave];
    #pragma unroll
    for (int nt = 0; nt < 8; ++nt)
        #pragma unroll
        for (int v = 0; v < 8; ++v) {
            int row = v + 8 * hi;
            long long nd = base + row;
            long long ndc = (nd < n_nodes) ? nd : (n_nodes - 1);
            obuf[row * HID + nt * 16 + r] =
                acc2[nt][v] + nf[ndc * HID + nt * 16 + r];
        }
    __syncthreads();

    // ---- layernorm: lanes 0..15 each own one row ----
    if (lane < 16) {
        long long nd = base + lane;
        if (nd < n_nodes) {
            const float* xb = &obuf[lane * HID];
            float mu = 0.0f;
            #pragma unroll 8
            for (int c = 0; c < HID; ++c) mu += xb[c];
            mu *= (1.0f / HID);
            float var = 0.0f;
            #pragma unroll 8
            for (int c = 0; c < HID; ++c) { float t = xb[c] - mu; var += t * t; }
            var *= (1.0f / HID);
            float rs = rsqrtf(var + 1e-5f);
            float* op = out + nd * HID;
            #pragma unroll 8
            for (int c = 0; c < HID; ++c)
                op[c] = (xb[c] - mu) * rs * gamma[c] + beta[c];
        }
    }
}

// ---------------------------------------------------------------------------
extern "C" void kernel_launch(void* const* d_in, const int* in_sizes, int n_in,
                              void* d_out, int out_size, void* d_ws, size_t ws_size,
                              hipStream_t stream) {
    const float*     nf    = (const float*)d_in[0];
    const long long* ei    = (const long long*)d_in[1];   // int64 per reference
    const float*     ef    = (const float*)d_in[2];
    const float*     coord = (const float*)d_in[3];
    const float*     mW1   = (const float*)d_in[4];
    const float*     mb1   = (const float*)d_in[5];
    const float*     mW2   = (const float*)d_in[6];
    const float*     mb2   = (const float*)d_in[7];
    const float*     uW1   = (const float*)d_in[8];
    const float*     ub1   = (const float*)d_in[9];
    const float*     uW2   = (const float*)d_in[10];
    const float*     ub2   = (const float*)d_in[11];
    const float*     lng   = (const float*)d_in[12];
    const float*     lnb   = (const float*)d_in[13];
    float*           out   = (float*)d_out;

    const int n_nodes = in_sizes[0] / HID;
    const int n_edges = in_sizes[1] / 2;

    // workspace layout (d_ws): packed f16 weights, then f32 aggregate buffer
    char* ws = (char*)d_ws;
    _Float16* W1p = (_Float16*)ws;
    _Float16* W2p = W1p + W1_H;
    _Float16* U1p = W2p + W2_H;
    _Float16* U2p = U1p + U1_H;
    float*    agg = (float*)(ws + 2 * (W1_H + W2_H + U1_H + U2_H)); // 221184B, 256-aligned

    // 1) pack weights into WMMA B-fragment layout (f16, K zero-padded)
    pack_w<<<(W1_H + 255) / 256, 256, 0, stream>>>(mW1, W1p, 290, W1_H);
    pack_w<<<(W2_H + 255) / 256, 256, 0, stream>>>(mW2, W2p, 128, W2_H);
    pack_w<<<(U1_H + 255) / 256, 256, 0, stream>>>(uW1, U1p, 257, U1_H);
    pack_w<<<(U2_H + 255) / 256, 256, 0, stream>>>(uW2, U2p, 128, U2_H);

    // 2) clear aggregate
    long long agg_n = (long long)n_nodes * HID;
    zero_f32<<<(unsigned)((agg_n + 255) / 256), 256, 0, stream>>>(agg, agg_n);

    // 3) edge message MLP + scatter (1 wave per 32 edges, 4 waves/block)
    int ewaves  = (n_edges + 31) / 32;
    int eblocks = (ewaves + 3) / 4;
    edge_msg<<<eblocks, 128, 0, stream>>>(nf, ei, ef, coord,
                                          W1p, mb1, W2p, mb2, agg, n_edges);

    // 4) node update MLP + residual + layernorm
    int nwaves  = (n_nodes + 15) / 16;
    int nblocks = (nwaves + 3) / 4;
    node_upd<<<nblocks, 128, 0, stream>>>(nf, agg, coord,
                                          U1p, ub1, U2p, ub2, lng, lnb,
                                          out, n_nodes);
}
